// DygRegressor_68788196213103
// MI455X (gfx1250) — matrix operand order repaired and checked by reading, not compiled, not run on hardware
//
#include <hip/hip_runtime.h>
#include <hip/hip_bf16.h>

typedef __attribute__((ext_vector_type(16))) _Float16 v16h;
typedef __attribute__((ext_vector_type(8)))  float    v8f;

#define N_ROWS    4000
#define T_STEPS   32
#define H_DIM     64
#define H3        192
#define E_EDGES   1024000
#define NT_NODES  (N_ROWS * T_STEPS)      // 128000
#define B_BATCH   8
#define NSTEPS    12

// Dynamic LDS layout for the GRU kernel (per block of 2 waves):
//   [0)      sWih  f16 swizzled   24576 B
//   [24576)  sWhh  f16 swizzled   24576 B
//   [49152)  sWg   f16 swizzled    8192 B
//   [57344)  sbih  f32              768 B
//   [58112)  sbhh  f32              768 B
//   [58880)  h-mirror f16, per wave 2048 B * 2
//   [62976)  x double-buffer f32, per wave 2*4096 B * 2
#define OFF_WIH   0
#define OFF_WHH   24576
#define OFF_WG    49152
#define OFF_BIH   57344
#define OFF_BHH   58112
#define OFF_HST   58880
#define OFF_XST   62976
#define SMEM_GRU  (62976 + 2 * 8192)      // 79360 B

// ---------------------------------------------------------------------------
// helpers
// ---------------------------------------------------------------------------
__device__ __forceinline__ void lds_fence() {
  // CDNA5 split dependency counter: order wave-synchronous LDS staging
  // against the WMMA-operand rebuild reads.
  asm volatile("s_wait_dscnt 0" ::: "memory");
}

__device__ __forceinline__ void async_fence() {
  asm volatile("s_wait_asynccnt 0" ::: "memory");
}

__device__ __forceinline__ v8f wmma_f16(v16h a, v16h b, v8f c) {
  // (neg_a, A, neg_b, B, c_mod, C, reuse_a, reuse_b)
  return __builtin_amdgcn_wmma_f32_16x16x32_f16(false, a, false, b, (short)0, c,
                                                false, false);
}

__device__ __forceinline__ float sigf(float x) {
  return 1.0f / (1.0f + __expf(-x));
}

// Build A operand (16x32 f16, M across lanes 0..15; lanes 0-15 hold
// K {0..7,16..23}, lanes 16-31 hold K {8..15,24..31}) from f16 [16][64] LDS.
__device__ __forceinline__ v16h build_a(const _Float16* st, int lane, int kchunk) {
  int m  = lane & 15;
  int kb = (lane & 16) ? 8 : 0;
  v16h a;
#pragma unroll
  for (int e = 0; e < 16; ++e) {
    int kk = (e < 8) ? e : (e + 8);
    a[e] = st[m * H_DIM + kchunk + kb + kk];
  }
  return a;
}

// Same A build, but source tile is f32 (async-copied raw from global).
__device__ __forceinline__ v16h build_a32(const float* st, int lane, int kchunk) {
  int m  = lane & 15;
  int kb = (lane & 16) ? 8 : 0;
  v16h a;
#pragma unroll
  for (int e = 0; e < 16; ++e) {
    int kk = (e < 8) ? e : (e + 8);
    a[e] = (_Float16)st[m * H_DIM + kchunk + kb + kk];
  }
  return a;
}

// B operand pre-swizzled in LDS: [ntile][kchunk][lane][16 halves] contiguous.
__device__ __forceinline__ v16h load_b(const _Float16* w, int jt, int kc, int lane) {
  const _Float16* p = w + (((jt * 2 + kc) * 32 + lane) << 4);
  v16h b;
#pragma unroll
  for (int e = 0; e < 16; ++e) b[e] = p[e];
  return b;
}

__device__ __forceinline__ v8f bias_vec(const float* b, int jt, int col) {
  float v = b[jt * 16 + col];
  v8f r = {v, v, v, v, v, v, v, v};
  return r;
}

// Async prefetch of one wave's 16x64 f32 x-tile for time step t into LDS.
// One row = 64 f32 = 256 B = 32 lanes * b64. GVS mode: SGPR base + v32 offset.
__device__ __forceinline__ void async_x_prefetch(const float* xblock, unsigned ldsbuf,
                                                 int t, int lane) {
  unsigned long long sbase = (unsigned long long)xblock;   // wave-uniform
#pragma unroll
  for (int r = 0; r < 16; ++r) {
    unsigned voff = (unsigned)(r * T_STEPS * H_DIM * 4 + t * H_DIM * 4 + lane * 8);
    unsigned ldst = ldsbuf + (unsigned)(r * 256 + lane * 8);
    asm volatile("global_load_async_to_lds_b64 %0, %1, %2"
                 :: "v"(ldst), "v"(voff), "s"(sbase)
                 : "memory");
  }
}

// ---------------------------------------------------------------------------
// Kernel 1: fused GRU (T=32 recurrence) + GCN input projection xw = h_t @ Wg^T
// 2 waves / block, each wave owns 16 rows for all 32 time steps.
// x tiles arrive via double-buffered GLOBAL_LOAD_ASYNC_TO_LDS (ASYNCcnt).
// ---------------------------------------------------------------------------
__global__ void __launch_bounds__(64)
dyg_gru_proj_kernel(const float* __restrict__ x, const float* __restrict__ Wih,
                    const float* __restrict__ Whh, const float* __restrict__ bih,
                    const float* __restrict__ bhh, const float* __restrict__ Wgcn,
                    float* __restrict__ xw) {
  extern __shared__ char smem[];
  _Float16* sWih = (_Float16*)(smem + OFF_WIH);
  _Float16* sWhh = (_Float16*)(smem + OFF_WHH);
  _Float16* sWg  = (_Float16*)(smem + OFF_WG);
  float*    sbih = (float*)(smem + OFF_BIH);
  float*    sbhh = (float*)(smem + OFF_BHH);

  const int tid = threadIdx.x;

  // Cooperative weight swizzle into B-operand layout: lane<16 -> N=lane,K={0..7,16..23} ;
  // lane>=16 -> N=lane-16,K={8..15,24..31} (per 32-wide K chunk). B[k][n] = W[n][k].
  for (int idx = tid; idx < 12 * 2 * 32 * 16; idx += 64) {
    int e = idx & 15, lane = (idx >> 4) & 31, kc = (idx >> 9) & 1, jt = idx >> 10;
    int n = jt * 16 + (lane & 15);
    int k = kc * 32 + ((lane & 16) ? 8 : 0) + ((e < 8) ? e : (e + 8));
    sWih[idx] = (_Float16)Wih[n * H_DIM + k];
    sWhh[idx] = (_Float16)Whh[n * H_DIM + k];
  }
  for (int idx = tid; idx < 4 * 2 * 32 * 16; idx += 64) {
    int e = idx & 15, lane = (idx >> 4) & 31, kc = (idx >> 9) & 1, jt = idx >> 10;
    int n = jt * 16 + (lane & 15);
    int k = kc * 32 + ((lane & 16) ? 8 : 0) + ((e < 8) ? e : (e + 8));
    sWg[idx] = (_Float16)Wgcn[n * H_DIM + k];
  }
  for (int idx = tid; idx < H3; idx += 64) {
    sbih[idx] = bih[idx];
    sbhh[idx] = bhh[idx];
  }
  __syncthreads();

  const int wave = tid >> 5, lane = tid & 31;
  _Float16* hst      = (_Float16*)(smem + OFF_HST + wave * 2048);
  const unsigned xoff0 = OFF_XST + wave * 8192;        // buffer 0
  const unsigned xoff1 = xoff0 + 4096;                 // buffer 1
  const int rowBase = blockIdx.x * 32 + wave * 16;     // 4000 % 32 == 0
  const float* xblock = x + (size_t)rowBase * T_STEPS * H_DIM;
  const int m8  = (lane & 16) ? 8 : 0;                 // C/D row offset
  const int col = lane & 15;                           // C/D column (N)

  // h0 = 0 : registers + LDS mirror; prime async prefetch of x[:,0,:].
  v8f h[4];
#pragma unroll
  for (int j = 0; j < 4; ++j) h[j] = (v8f){0, 0, 0, 0, 0, 0, 0, 0};
  for (int i = lane; i < 16 * H_DIM; i += 32) hst[i] = (_Float16)0.0f;
  async_x_prefetch(xblock, xoff0, 0, lane);
  lds_fence();

  for (int t = 0; t < T_STEPS; ++t) {
    // A operands of h_{t-1} (from LDS mirror)
    v16h ah0 = build_a(hst, lane, 0);
    v16h ah1 = build_a(hst, lane, 32);

    // Wait for async x tile of this step, build A operands, then immediately
    // kick off the prefetch of the next step's tile into the other buffer.
    async_fence();
    const float* xs = (const float*)(smem + xoff0 + ((t & 1) ? 4096u : 0u));
    v16h ax0 = build_a32(xs, lane, 0);
    v16h ax1 = build_a32(xs, lane, 32);
    if (t + 1 < T_STEPS)
      async_x_prefetch(xblock, (t & 1) ? xoff0 : xoff1, t + 1, lane);

    // Gate groups: tiles j (r), j+4 (z), j+8 (n) of the 16x192 projections.
#pragma unroll
    for (int j = 0; j < 4; ++j) {
      v8f xr = bias_vec(sbih, j, col),     hr = bias_vec(sbhh, j, col);
      v8f xz = bias_vec(sbih, j + 4, col), hz = bias_vec(sbhh, j + 4, col);
      v8f xn = bias_vec(sbih, j + 8, col), hn = bias_vec(sbhh, j + 8, col);

      xr = wmma_f16(ax0, load_b(sWih, j, 0, lane), xr);
      xr = wmma_f16(ax1, load_b(sWih, j, 1, lane), xr);
      hr = wmma_f16(ah0, load_b(sWhh, j, 0, lane), hr);
      hr = wmma_f16(ah1, load_b(sWhh, j, 1, lane), hr);
      xz = wmma_f16(ax0, load_b(sWih, j + 4, 0, lane), xz);
      xz = wmma_f16(ax1, load_b(sWih, j + 4, 1, lane), xz);
      hz = wmma_f16(ah0, load_b(sWhh, j + 4, 0, lane), hz);
      hz = wmma_f16(ah1, load_b(sWhh, j + 4, 1, lane), hz);
      xn = wmma_f16(ax0, load_b(sWih, j + 8, 0, lane), xn);
      xn = wmma_f16(ax1, load_b(sWih, j + 8, 1, lane), xn);
      hn = wmma_f16(ah0, load_b(sWhh, j + 8, 0, lane), hn);
      hn = wmma_f16(ah1, load_b(sWhh, j + 8, 1, lane), hn);

      v8f hnew;
#pragma unroll
      for (int v = 0; v < 8; ++v) {
        float rg = sigf(xr[v] + hr[v]);
        float zg = sigf(xz[v] + hz[v]);
        float ng = tanhf(xn[v] + rg * hn[v]);
        hnew[v]  = (1.0f - zg) * ng + zg * h[j][v];
      }
      h[j] = hnew;
    }

    // Mirror h_t into LDS (row-major f16 [16][64]) for next step + GCN A build.
#pragma unroll
    for (int j = 0; j < 4; ++j)
#pragma unroll
      for (int v = 0; v < 8; ++v)
        hst[(v + m8) * H_DIM + j * 16 + col] = (_Float16)h[j][v];
    lds_fence();

    // Fused GCN input projection: xw[nt, :] = h_t @ Wgcn^T  (nt = row*T + t)
    v16h ag0 = build_a(hst, lane, 0);
    v16h ag1 = build_a(hst, lane, 32);
#pragma unroll
    for (int jt = 0; jt < 4; ++jt) {
      v8f acc = (v8f){0, 0, 0, 0, 0, 0, 0, 0};
      acc = wmma_f16(ag0, load_b(sWg, jt, 0, lane), acc);
      acc = wmma_f16(ag1, load_b(sWg, jt, 1, lane), acc);
#pragma unroll
      for (int v = 0; v < 8; ++v) {
        int row = rowBase + v + m8;
        xw[((size_t)row * T_STEPS + t) * H_DIM + jt * 16 + col] = acc[v];
      }
    }
  }
}

// ---------------------------------------------------------------------------
// GCN normalization + scatter kernels (working set ~70 MB: lives in 192MB L2)
// ---------------------------------------------------------------------------
__global__ void deg_init_kernel(float* __restrict__ deg, int n) {
  int i = blockIdx.x * blockDim.x + threadIdx.x;
  if (i < n) deg[i] = 1.0f;   // self-loop weight
}

__global__ void deg_scatter_kernel(const int* __restrict__ dst,
                                   const float* __restrict__ w,
                                   float* __restrict__ deg, int e) {
  int i = blockIdx.x * blockDim.x + threadIdx.x;
  if (i < e) atomicAdd(&deg[dst[i]], w[i]);
}

__global__ void dis_kernel(float* __restrict__ deg, int n) {
  int i = blockIdx.x * blockDim.x + threadIdx.x;
  if (i < n) deg[i] = rsqrtf(deg[i]);   // deg >= 1 always (self-loop)
}

__global__ void agg_init_kernel(const float* __restrict__ xw,
                                const float* __restrict__ dis,
                                float* __restrict__ agg, int total) {
  int i = blockIdx.x * blockDim.x + threadIdx.x;
  if (i < total) {
    int nt = i >> 6;
    float d = dis[nt];
    agg[i] = d * d * xw[i];   // self-loop contribution (w = 1)
  }
}

__global__ void edge_scatter_kernel(const int* __restrict__ src,
                                    const int* __restrict__ dst,
                                    const float* __restrict__ w,
                                    const float* __restrict__ dis,
                                    const float* __restrict__ xw,
                                    float* __restrict__ agg, int total) {
  int gid = blockIdx.x * blockDim.x + threadIdx.x;
  if (gid >= total) return;
  int e = gid >> 6, c = gid & 63;
  int s = src[e], d = dst[e];
  float nrm = dis[s] * w[e] * dis[d];
  atomicAdd(&agg[(size_t)d * H_DIM + c], nrm * xw[(size_t)s * H_DIM + c]);
}

// ---------------------------------------------------------------------------
// Attention pre-reduction: per (b,t): 64 column sums over the contiguous
// 32000-float chunk (torch .view semantics) + chunk mean -> xt.
// ---------------------------------------------------------------------------
__global__ void __launch_bounds__(64)
colsum_kernel(const float* __restrict__ agg, float* __restrict__ csum,
              float* __restrict__ xt) {
  int bt = blockIdx.x;                    // b*32 + t, 0..255
  int hh = threadIdx.x;                   // 0..63
  const float* base = agg + (size_t)bt * 32000;
  float s = 0.0f;
  for (int i = 0; i < 500; ++i) s += base[i * 64 + hh];
  csum[bt * 64 + hh] = s;
  __shared__ float red[64];
  red[hh] = s;
  __syncthreads();
  if (hh == 0) {
    float tot = 0.0f;
    for (int i = 0; i < 64; ++i) tot += red[i];
    xt[bt] = tot * (1.0f / 32000.0f);
  }
}

// ---------------------------------------------------------------------------
// Attention MLP + pooling + final head (tiny, one block)
// ---------------------------------------------------------------------------
__global__ void __launch_bounds__(256)
head_kernel(const float* __restrict__ xt, const float* __restrict__ csum,
            const float* __restrict__ W1, const float* __restrict__ W2,
            const float* __restrict__ fcW, const float* __restrict__ fcb,
            float* __restrict__ out) {
  __shared__ float sxt[256];
  __shared__ float sattn[256];
  __shared__ float spooled[512];
  int tid = threadIdx.x;
  sxt[tid] = xt[tid];
  __syncthreads();
  {
    int b = tid >> 5, t = tid & 31;
    float acc = 0.0f;
    for (int k = 0; k < 16; ++k) {
      float hk = 0.0f;
      for (int tt = 0; tt < 32; ++tt) hk += sxt[b * 32 + tt] * W1[k * 32 + tt];
      hk = fmaxf(hk, 0.0f);
      acc += hk * W2[t * 16 + k];
    }
    sattn[tid] = 1.0f / (1.0f + __expf(-acc));
  }
  __syncthreads();
  for (int idx = tid; idx < 512; idx += 256) {
    int b = idx >> 6, hh = idx & 63;
    float p = 0.0f;
    for (int t = 0; t < 32; ++t) p += sattn[b * 32 + t] * csum[(b * 32 + t) * 64 + hh];
    spooled[idx] = p;
  }
  __syncthreads();
  if (tid < 96) {
    int b = tid / 12, s = tid % 12;
    float o = fcb[s];
    for (int hh = 0; hh < 64; ++hh) o += spooled[b * 64 + hh] * fcW[s * 64 + hh];
    out[b * 12 + s] = o;
  }
}

// ---------------------------------------------------------------------------
extern "C" void kernel_launch(void* const* d_in, const int* in_sizes, int n_in,
                              void* d_out, int out_size, void* d_ws, size_t ws_size,
                              hipStream_t stream) {
  const float* x   = (const float*)d_in[0];
  const int*   ei  = (const int*)d_in[1];
  const float* ew  = (const float*)d_in[2];
  // d_in[3] = batch (unused by the reference computation)
  const float* Wih = (const float*)d_in[4];
  const float* Whh = (const float*)d_in[5];
  const float* bih = (const float*)d_in[6];
  const float* bhh = (const float*)d_in[7];
  const float* Wg  = (const float*)d_in[8];
  const float* W1  = (const float*)d_in[9];
  const float* W2  = (const float*)d_in[10];
  const float* fcW = (const float*)d_in[11];
  const float* fcb = (const float*)d_in[12];

  const int* src = ei;             // edge_index[0]
  const int* dst = ei + E_EDGES;   // edge_index[1]

  char* ws = (char*)d_ws;
  float* xw   = (float*)(ws);                       // NT*64 f32 = 32,768,000 B
  float* agg  = (float*)(ws + 32768000);            // NT*64 f32 = 32,768,000 B
  float* deg  = (float*)(ws + 65536000);            // NT f32 (reused as dis)
  float* csum = (float*)(ws + 66048000);            // 256*64 f32
  float* xt   = (float*)(ws + 66113536);            // 256 f32
  float* out  = (float*)d_out;

  const int NT = NT_NODES;
  const int totalAgg  = NT * H_DIM;                 // 8,192,000
  const int totalEdge = E_EDGES * H_DIM;            // 65,536,000

  dyg_gru_proj_kernel<<<N_ROWS / 32, 64, SMEM_GRU, stream>>>(x, Wih, Whh, bih,
                                                             bhh, Wg, xw);
  deg_init_kernel<<<(NT + 255) / 256, 256, 0, stream>>>(deg, NT);
  deg_scatter_kernel<<<(E_EDGES + 255) / 256, 256, 0, stream>>>(dst, ew, deg, E_EDGES);
  dis_kernel<<<(NT + 255) / 256, 256, 0, stream>>>(deg, NT);
  agg_init_kernel<<<(totalAgg + 255) / 256, 256, 0, stream>>>(xw, deg, agg, totalAgg);
  edge_scatter_kernel<<<(totalEdge + 255) / 256, 256, 0, stream>>>(src, dst, ew, deg,
                                                                   xw, agg, totalEdge);
  colsum_kernel<<<256, 64, 0, stream>>>(agg, csum, xt);
  head_kernel<<<1, 256, 0, stream>>>(xt, csum, W1, W2, fcW, fcb, out);
}
